// TokenGATClassifier_88776974008768
// MI455X (gfx1250) — matrix operand order repaired and checked by reading, not compiled, not run on hardware
//
#include <hip/hip_runtime.h>
#include <hip/hip_bf16.h>

// ---------------- problem constants (match reference) ----------------
#define NNODES  50000
#define NEDGES  400000
#define NGRAPH  64
#define DH      128
#define NHEAD   4
#define NLAYER  3
#define EPLUS   (NEDGES + NNODES)   // 450000, multiple of 16

typedef __bf16 bf16;
typedef __attribute__((ext_vector_type(16))) __bf16 v16bf;
typedef __attribute__((ext_vector_type(8)))  float  v8f;

union ABu { v16bf v; unsigned u[8]; };

__device__ __forceinline__ unsigned short f2bf(float f) {
  unsigned u = __float_as_uint(f);
  unsigned r = u + 0x7FFFu + ((u >> 16) & 1u);   // round-to-nearest-even
  return (unsigned short)(r >> 16);
}
__device__ __forceinline__ unsigned encf(float f) {   // order-preserving encode
  unsigned u = __float_as_uint(f);
  return (u & 0x80000000u) ? ~u : (u | 0x80000000u);
}
__device__ __forceinline__ float decf(unsigned e) {
  unsigned u = (e & 0x80000000u) ? (e & 0x7FFFFFFFu) : ~e;
  return __uint_as_float(u);
}
__device__ __forceinline__ void atomAddF(float* p, float v) {
  __hip_atomic_fetch_add(p, v, __ATOMIC_RELAXED, __HIP_MEMORY_SCOPE_AGENT);
}
__device__ __forceinline__ void atomMaxU(unsigned* p, unsigned v) {
  __hip_atomic_fetch_max(p, v, __ATOMIC_RELAXED, __HIP_MEMORY_SCOPE_AGENT);
}

// ---------------- weight pre-pack into WMMA B-lane layout ----------------
// Bpack dword index: (((ntile*ksteps)+s)*32 + lane)*8 + j
// holds bf16 pair (K=kbase, K=kbase+1) of column (ntile*16 + lane&15),
// kbase = s*32 + (j>>2)*16 + (lane>>4)*8 + 2*(j&3)   (16-bit B 32x16 wave32 layout)
__global__ void pack_w(const float* __restrict__ W, unsigned* __restrict__ P, int K) {
  int ksteps = K >> 5;
  int total  = 8 * ksteps * 32 * 8;
  int idx = blockIdx.x * blockDim.x + threadIdx.x;
  if (idx >= total) return;
  int j    = idx & 7;
  int lane = (idx >> 3) & 31;
  int ts   = idx >> 8;
  int s    = ts % ksteps;
  int t    = ts / ksteps;
  int krow = s * 32 + ((j >> 2) << 4) + ((lane >> 4) << 3) + ((j & 3) << 1);
  int col  = t * 16 + (lane & 15);
  unsigned short lo = f2bf(W[(size_t)krow * DH + col]);
  unsigned short hi = f2bf(W[(size_t)(krow + 1) * DH + col]);
  P[idx] = (unsigned)lo | ((unsigned)hi << 16);
}

// ---------------- generic WMMA GEMM: Out(M,128) = A(M,K)*B + bias ----------------
// 64 rows per block (4 subtiles of 16); B held in registers across subtiles.
// K is a compile-time constant so breg[] indexing is fully static (no scratch spill).
template <int K>
__global__ __launch_bounds__(256)
void gemm_bias_t(const float* __restrict__ A, const unsigned* __restrict__ Bp,
                 const float* __restrict__ bias, float* __restrict__ Out,
                 int do_relu, int M) {
  constexpr int KSTEPS = K >> 5;           // 2 or 4
  constexpr int KSHIFT = (K == 64) ? 6 : 7;
  __shared__ unsigned short As[64][K];     // <= 16 KB
  const int tid  = threadIdx.x;
  const int wave = tid >> 5;
  const int lane = tid & 31;
  const int rowbase = blockIdx.x * 64;
  // B fragments for this wave's 16-col tile, all k-steps, in registers
  ABu breg[KSTEPS];
#pragma unroll
  for (int s = 0; s < KSTEPS; ++s) {
    const unsigned* bp = Bp + (((size_t)(wave * KSTEPS + s)) * 32 + lane) * 8;
#pragma unroll
    for (int j = 0; j < 8; ++j) breg[s].u[j] = bp[j];
  }
  // stage A (64 rows x K) as bf16, zero-pad past M
  for (int i = tid; i < 64 * K; i += 256) {
    int r = i >> KSHIFT, c = i & (K - 1);
    int gr = rowbase + r;
    As[r][c] = (gr < M) ? f2bf(A[(size_t)gr * K + c]) : (unsigned short)0;
  }
  __syncthreads();
  const int col = wave * 16 + (lane & 15);
  const float bv = bias[col];
  const int kh = lane >> 4;
#pragma unroll
  for (int sub = 0; sub < 4; ++sub) {
    v8f acc = {};
    const int m = sub * 16 + (lane & 15);
#pragma unroll
    for (int s = 0; s < KSTEPS; ++s) {
      ABu a;
#pragma unroll
      for (int j = 0; j < 8; ++j) {
        int kk = s * 32 + ((j >> 2) << 4) + ((j & 3) << 1) + kh * 8;
        a.u[j] = *(const unsigned*)&As[m][kk];
      }
      acc = __builtin_amdgcn_wmma_f32_16x16x32_bf16(false, a.v, false, breg[s].v,
                                                    (short)0, acc, false, false);
    }
#pragma unroll
    for (int j = 0; j < 8; ++j) {
      int r = rowbase + sub * 16 + (lane >> 4) * 8 + j;
      if (r < M) {
        float v = acc[j] + bv;
        if (do_relu) v = v > 0.f ? v : 0.f;
        Out[(size_t)r * DH + col] = v;
      }
    }
  }
}

// ---------------- fused e = ef_all@We GEMM + attention-logit epilogue ----------------
// 64 rows per block; B in registers; per-subtile shuffle+LDS head reduction.
__global__ __launch_bounds__(256)
void gemm_edge_logit(const float* __restrict__ A, const unsigned* __restrict__ Bp,
                     const float* __restrict__ xl, const float* __restrict__ xr,
                     const int* __restrict__ src, const int* __restrict__ dst,
                     const float* __restrict__ attw, float* __restrict__ logit,
                     int E, int M) {
  __shared__ unsigned short As[64][128];   // 16 KB
  __shared__ int rs[64], rd[64];
  __shared__ float part[8][16];
  const int tid  = threadIdx.x;
  const int wave = tid >> 5;
  const int lane = tid & 31;
  const int rowbase = blockIdx.x * 64;
  if (tid < 64) {
    int gr = rowbase + tid;
    int ok = gr < M;
    rs[tid] = ok ? (gr < E ? src[gr] : (gr - E)) : 0;
    rd[tid] = ok ? (gr < E ? dst[gr] : (gr - E)) : 0;
  }
  ABu breg[4];
#pragma unroll
  for (int s = 0; s < 4; ++s) {
    const unsigned* bp = Bp + (((size_t)(wave * 4 + s)) * 32 + lane) * 8;
#pragma unroll
    for (int j = 0; j < 8; ++j) breg[s].u[j] = bp[j];
  }
  for (int i = tid; i < 64 * 128; i += 256) {
    int r = i >> 7, c = i & 127;
    int gr = rowbase + r;
    As[r][c] = (gr < M) ? f2bf(A[(size_t)gr * 128 + c]) : (unsigned short)0;
  }
  __syncthreads();
  const int col = wave * 16 + (lane & 15);
  const float aw = attw[col];
  const int kh = lane >> 4;
#pragma unroll
  for (int sub = 0; sub < 4; ++sub) {
    v8f acc = {};
    const int m = sub * 16 + (lane & 15);
#pragma unroll
    for (int s = 0; s < 4; ++s) {
      ABu a;
#pragma unroll
      for (int j = 0; j < 8; ++j) {
        int kk = s * 32 + ((j >> 2) << 4) + ((j & 3) << 1) + kh * 8;
        a.u[j] = *(const unsigned*)&As[m][kk];
      }
      acc = __builtin_amdgcn_wmma_f32_16x16x32_bf16(false, a.v, false, breg[s].v,
                                                    (short)0, acc, false, false);
    }
    // m = leaky_relu(xl[src]+xr[dst]+e); partial logit = m * att[col]
    float p8[8];
#pragma unroll
    for (int j = 0; j < 8; ++j) {
      int rl = sub * 16 + (lane >> 4) * 8 + j;
      float mv = acc[j] + xl[(size_t)rs[rl] * DH + col] + xr[(size_t)rd[rl] * DH + col];
      mv = mv > 0.f ? mv : 0.2f * mv;
      p8[j] = mv * aw;
    }
#pragma unroll
    for (int d = 1; d < 16; d <<= 1) {
#pragma unroll
      for (int j = 0; j < 8; ++j) p8[j] += __shfl_xor(p8[j], d, 32);
    }
    if ((lane & 15) == 0) {
#pragma unroll
      for (int j = 0; j < 8; ++j) part[wave][(lane >> 4) * 8 + j] = p8[j];
    }
    __syncthreads();
    if (tid < 64) {   // wave w covers cols w*16..w*16+15 -> head = w>>1
      int row = tid & 15, head = tid >> 4;
      int gr = rowbase + sub * 16 + row;
      if (gr < M)
        logit[(size_t)gr * NHEAD + head] = part[2 * head][row] + part[2 * head + 1][row];
    }
    __syncthreads();   // protect part[] before next subtile overwrites
  }
}

// ---------------- small elementwise / scatter kernels ----------------
__global__ void fill_f32(float* p, int n, float v) {
  int i = blockIdx.x * blockDim.x + threadIdx.x;
  if (i < n) p[i] = v;
}
__global__ void fill_u32(unsigned* p, int n, unsigned v) {
  int i = blockIdx.x * blockDim.x + threadIdx.x;
  if (i < n) p[i] = v;
}
// per-node mean of incoming edge features (self-loop feature fill)
__global__ void scatter_ef(const float* __restrict__ ef, const int* __restrict__ dst,
                           float* __restrict__ sums, float* __restrict__ cnt) {
  int idx = blockIdx.x * blockDim.x + threadIdx.x;   // NEDGES*32
  if (idx >= NEDGES * 32) return;
  int e = idx >> 5, c0 = (idx & 31) * 4;
  int d = dst[e];
  const float* s = ef + (size_t)e * DH + c0;
  float* o = sums + (size_t)d * DH + c0;
  atomAddF(o + 0, s[0]); atomAddF(o + 1, s[1]);
  atomAddF(o + 2, s[2]); atomAddF(o + 3, s[3]);
  if (c0 == 0) atomAddF(&cnt[d], 1.0f);
}
__global__ void div_mean(float* __restrict__ sums, const float* __restrict__ cnt) {
  int i = blockIdx.x * blockDim.x + threadIdx.x;     // NNODES*128
  if (i >= NNODES * DH) return;
  sums[i] /= fmaxf(cnt[i >> 7], 1.0f);
}
__global__ void seg_max(const float* __restrict__ logit, const int* __restrict__ dst,
                        unsigned* __restrict__ mx, int E) {
  int idx = blockIdx.x * blockDim.x + threadIdx.x;   // EPLUS*4
  if (idx >= EPLUS * NHEAD) return;
  int row = idx >> 2, h = idx & 3;
  int d = row < E ? dst[row] : (row - E);
  atomMaxU(&mx[d * NHEAD + h], encf(logit[idx]));
}
__global__ void seg_expden(float* __restrict__ logit, const int* __restrict__ dst,
                           const unsigned* __restrict__ mx, float* __restrict__ den, int E) {
  int idx = blockIdx.x * blockDim.x + threadIdx.x;   // EPLUS*4
  if (idx >= EPLUS * NHEAD) return;
  int row = idx >> 2, h = idx & 3;
  int d = row < E ? dst[row] : (row - E);
  float ex = __expf(logit[idx] - decf(mx[d * NHEAD + h]));
  logit[idx] = ex;
  atomAddF(&den[d * NHEAD + h], ex);
}
__global__ void init_out(float* __restrict__ out, const float* __restrict__ bias) {
  int i = blockIdx.x * blockDim.x + threadIdx.x;     // NNODES*128
  if (i < NNODES * DH) out[i] = bias[i & 127];
}
__global__ void scatter_msg(const float* __restrict__ ex, const float* __restrict__ den,
                            const float* __restrict__ xl, const int* __restrict__ src,
                            const int* __restrict__ dst, float* __restrict__ out, int E) {
  int idx = blockIdx.x * blockDim.x + threadIdx.x;   // EPLUS*32
  if (idx >= EPLUS * 32) return;
  int e = idx >> 5, c0 = (idx & 31) * 4, head = c0 >> 5;
  int s_ = e < E ? src[e] : (e - E);
  int d_ = e < E ? dst[e] : (e - E);
  float alpha = ex[(size_t)e * NHEAD + head] / den[d_ * NHEAD + head];
  const float* xs = xl + (size_t)s_ * DH + c0;
  float* o = out + (size_t)d_ * DH + c0;
  atomAddF(o + 0, xs[0] * alpha); atomAddF(o + 1, xs[1] * alpha);
  atomAddF(o + 2, xs[2] * alpha); atomAddF(o + 3, xs[3] * alpha);
}
__global__ void finalize_bn(const float* __restrict__ out, const float* __restrict__ rmean,
                            const float* __restrict__ rvar, const float* __restrict__ gamma,
                            const float* __restrict__ beta, float* __restrict__ h, int first) {
  int i = blockIdx.x * blockDim.x + threadIdx.x;     // NNODES*128
  if (i >= NNODES * DH) return;
  int c = i & 127;
  float v = (out[i] - rmean[c]) * rsqrtf(rvar[c] + 1e-5f) * gamma[c] + beta[c];
  v = fmaxf(v, 0.f);
  h[i] = first ? v : (h[i] + v);
}
__global__ void pool_scatter(const float* __restrict__ h, const int* __restrict__ batch,
                             float* __restrict__ g, float* __restrict__ gcnt) {
  int idx = blockIdx.x * blockDim.x + threadIdx.x;   // NNODES*32
  if (idx >= NNODES * 32) return;
  int n = idx >> 5, c0 = (idx & 31) * 4;
  int b = batch[n];
  const float* s = h + (size_t)n * DH + c0;
  float* o = g + (size_t)b * DH + c0;
  atomAddF(o + 0, s[0]); atomAddF(o + 1, s[1]);
  atomAddF(o + 2, s[2]); atomAddF(o + 3, s[3]);
  if (c0 == 0) atomAddF(&gcnt[b], 1.0f);
}
__global__ __launch_bounds__(128)
void mlp_head(const float* __restrict__ g, const float* __restrict__ gcnt,
              const float* __restrict__ Wc1, const float* __restrict__ bc1,
              const float* __restrict__ Wc2, const float* __restrict__ bc2,
              const float* __restrict__ Wc3, const float* __restrict__ bc3,
              float* __restrict__ out) {
  __shared__ float gl[128], z1[128], z2[64];
  int gi = blockIdx.x, t = threadIdx.x;
  gl[t] = g[(size_t)gi * DH + t] / fmaxf(gcnt[gi], 1.0f);
  __syncthreads();
  float a = bc1[t];
  for (int k = 0; k < 128; ++k) a += gl[k] * Wc1[k * 128 + t];
  z1[t] = fmaxf(a, 0.f);
  __syncthreads();
  if (t < 64) {
    float b = bc2[t];
    for (int k = 0; k < 128; ++k) b += z1[k] * Wc2[k * 64 + t];
    z2[t] = fmaxf(b, 0.f);
  }
  __syncthreads();
  if (t < 2) {
    float c = bc3[t];
    for (int k = 0; k < 64; ++k) c += z2[k] * Wc3[k * 2 + t];
    out[gi * 2 + t] = c;
  }
}

// ---------------- host orchestration ----------------
extern "C" void kernel_launch(void* const* d_in, const int* in_sizes, int n_in,
                              void* d_out, int out_size, void* d_ws, size_t ws_size,
                              hipStream_t stream) {
  const float* x        = (const float*)d_in[0];
  const float* edge_attr= (const float*)d_in[1];
  const float* W_in     = (const float*)d_in[2];
  const float* b_in     = (const float*)d_in[3];
  const float* W_edge   = (const float*)d_in[4];
  const float* b_edge   = (const float*)d_in[5];
  const float* Wl       = (const float*)d_in[6];
  const float* bl       = (const float*)d_in[7];
  const float* Wr       = (const float*)d_in[8];
  const float* br       = (const float*)d_in[9];
  const float* We       = (const float*)d_in[10];
  const float* att      = (const float*)d_in[11];
  const float* bias_out = (const float*)d_in[12];
  const float* gamma    = (const float*)d_in[13];
  const float* beta     = (const float*)d_in[14];
  const float* rmean    = (const float*)d_in[15];
  const float* rvar     = (const float*)d_in[16];
  const float* Wc1      = (const float*)d_in[17];
  const float* bc1      = (const float*)d_in[18];
  const float* Wc2      = (const float*)d_in[19];
  const float* bc2      = (const float*)d_in[20];
  const float* Wc3      = (const float*)d_in[21];
  const float* bc3      = (const float*)d_in[22];
  const int*   eidx     = (const int*)d_in[23];
  const int*   batch    = (const int*)d_in[24];
  const int*   srcp = eidx;
  const int*   dstp = eidx + NEDGES;

  char* base = (char*)d_ws;
  size_t off = 0;
  auto carve = [&](size_t bytes) -> char* {
    char* p = base + off;
    off += (bytes + 255) & ~(size_t)255;
    return p;
  };
  float*    hbuf   = (float*)   carve((size_t)NNODES * DH * 4);
  float*    efall  = (float*)   carve((size_t)EPLUS  * DH * 4);
  float*    xlb    = (float*)   carve((size_t)NNODES * DH * 4);
  float*    xrb    = (float*)   carve((size_t)NNODES * DH * 4);
  float*    logitb = (float*)   carve((size_t)EPLUS  * NHEAD * 4);
  unsigned* mx     = (unsigned*)carve((size_t)NNODES * NHEAD * 4);
  float*    den    = (float*)   carve((size_t)NNODES * NHEAD * 4);
  float*    outacc = (float*)   carve((size_t)NNODES * DH * 4);
  float*    cnt    = (float*)   carve((size_t)NNODES * 4);
  float*    g      = (float*)   carve((size_t)NGRAPH * DH * 4);
  float*    gcnt   = (float*)   carve((size_t)NGRAPH * 4);
  unsigned* pWin   = (unsigned*)carve(8192 * 4);
  unsigned* pWedge = (unsigned*)carve(4096 * 4);
  unsigned* pWl[NLAYER]; unsigned* pWr[NLAYER]; unsigned* pWe[NLAYER];
  for (int i = 0; i < NLAYER; ++i) {
    pWl[i] = (unsigned*)carve(8192 * 4);
    pWr[i] = (unsigned*)carve(8192 * 4);
    pWe[i] = (unsigned*)carve(8192 * 4);
  }

  // 1) pack weights to WMMA layout (bf16)
  pack_w<<<32, 256, 0, stream>>>(W_in, pWin, 128);
  pack_w<<<16, 256, 0, stream>>>(W_edge, pWedge, 64);
  for (int i = 0; i < NLAYER; ++i) {
    pack_w<<<32, 256, 0, stream>>>(Wl + (size_t)i * 128 * 128, pWl[i], 128);
    pack_w<<<32, 256, 0, stream>>>(Wr + (size_t)i * 128 * 128, pWr[i], 128);
    pack_w<<<32, 256, 0, stream>>>(We + (size_t)i * 128 * 128, pWe[i], 128);
  }

  // 2) h = relu(x @ W_in + b_in);  ef = edge_attr @ W_edge + b_edge
  gemm_bias_t<128><<<(NNODES + 63) / 64, 256, 0, stream>>>(x, pWin, b_in, hbuf, 1, NNODES);
  gemm_bias_t<64><<<(NEDGES + 63) / 64, 256, 0, stream>>>(edge_attr, pWedge, b_edge, efall, 0, NEDGES);

  // 3) self-loop features = per-node mean of incoming ef
  fill_f32<<<(NNODES * DH) / 256, 256, 0, stream>>>(efall + (size_t)NEDGES * DH, NNODES * DH, 0.f);
  fill_f32<<<(NNODES + 255) / 256, 256, 0, stream>>>(cnt, NNODES, 0.f);
  scatter_ef<<<(NEDGES * 32) / 256, 256, 0, stream>>>(efall, dstp, efall + (size_t)NEDGES * DH, cnt);
  div_mean<<<(NNODES * DH) / 256, 256, 0, stream>>>(efall + (size_t)NEDGES * DH, cnt);

  // 4) GAT layers
  for (int i = 0; i < NLAYER; ++i) {
    gemm_bias_t<128><<<(NNODES + 63) / 64, 256, 0, stream>>>(hbuf, pWl[i], bl + i * DH, xlb, 0, NNODES);
    gemm_bias_t<128><<<(NNODES + 63) / 64, 256, 0, stream>>>(hbuf, pWr[i], br + i * DH, xrb, 0, NNODES);
    fill_u32<<<(NNODES * NHEAD + 255) / 256, 256, 0, stream>>>(mx, NNODES * NHEAD, 0u);
    fill_f32<<<(NNODES * NHEAD + 255) / 256, 256, 0, stream>>>(den, NNODES * NHEAD, 0.f);
    gemm_edge_logit<<<(EPLUS + 63) / 64, 256, 0, stream>>>(efall, pWe[i], xlb, xrb, srcp, dstp,
                                                           att + i * DH, logitb, NEDGES, EPLUS);
    seg_max<<<(EPLUS * NHEAD + 255) / 256, 256, 0, stream>>>(logitb, dstp, mx, NEDGES);
    seg_expden<<<(EPLUS * NHEAD + 255) / 256, 256, 0, stream>>>(logitb, dstp, mx, den, NEDGES);
    init_out<<<(NNODES * DH) / 256, 256, 0, stream>>>(outacc, bias_out + i * DH);
    scatter_msg<<<(EPLUS * 32) / 256, 256, 0, stream>>>(logitb, den, xlb, srcp, dstp, outacc, NEDGES);
    finalize_bn<<<(NNODES * DH) / 256, 256, 0, stream>>>(outacc, rmean + i * DH, rvar + i * DH,
                                                         gamma + i * DH, beta + i * DH, hbuf,
                                                         i == 0 ? 1 : 0);
  }

  // 5) mean pool per graph + MLP head
  fill_f32<<<(NGRAPH * DH + 255) / 256, 256, 0, stream>>>(g, NGRAPH * DH, 0.f);
  fill_f32<<<1, 256, 0, stream>>>(gcnt, NGRAPH, 0.f);
  pool_scatter<<<(NNODES * 32) / 256, 256, 0, stream>>>(hbuf, batch, g, gcnt);
  mlp_head<<<NGRAPH, 128, 0, stream>>>(g, gcnt, Wc1, bc1, Wc2, bc2, Wc3, bc3, (float*)d_out);
}
